// GraphSage_36550171689307
// MI455X (gfx1250) — matrix operand order repaired and checked by reading, not compiled, not run on hardware
//
#include <hip/hip_runtime.h>

typedef __bf16 bf16;
typedef __attribute__((ext_vector_type(16))) __bf16 v16bf;
typedef __attribute__((ext_vector_type(8)))  float   v8f;

#define N2_C  40960
#define R2_C  4096
#define R1_C  1024
#define S1_C  8192

// ---------------------------------------------------------------------------
// Gather rows of X (f32, [*,128]) into a transposed bf16 panel Bt[128][nJ].
// Consecutive j -> consecutive lanes, so the 2-byte stores coalesce per f.
// idx = sel[j], optionally composed through outer[] (features path).
// ---------------------------------------------------------------------------
__global__ void gather_bt(const int* __restrict__ sel, const int* __restrict__ outer,
                          const float* __restrict__ X, bf16* __restrict__ Bt, int nJ) {
  int j = blockIdx.x * blockDim.x + threadIdx.x;
  if (j >= nJ) return;
  int idx = sel[j];
  if (outer) idx = outer[idx];
  const float4* src = (const float4*)(X + (size_t)idx * 128);
#pragma unroll 4
  for (int q = 0; q < 32; ++q) {
    float4 v = src[q];
    size_t f = (size_t)(4 * q);
    Bt[(f + 0) * nJ + j] = (bf16)v.x;
    Bt[(f + 1) * nJ + j] = (bf16)v.y;
    Bt[(f + 2) * nJ + j] = (bf16)v.z;
    Bt[(f + 3) * nJ + j] = (bf16)v.w;
  }
}

// Gather dst rows (f32) into columns [128..256) of a [R x 256] concat matrix.
__global__ void gather_dst(const int* __restrict__ sel, const int* __restrict__ outer,
                           const float* __restrict__ X, float* __restrict__ Aout) {
  int r = blockIdx.x;
  int lane = threadIdx.x; // 32 threads
  int idx = sel[r];
  if (outer) idx = outer[idx];
  ((float4*)(Aout + (size_t)r * 256 + 128))[lane] =
      ((const float4*)(X + (size_t)idx * 128))[lane];
}

// Transpose dense weight w[K x 128] f32 -> Bt[128 x K] bf16.
__global__ void transpose_w(const float* __restrict__ w, bf16* __restrict__ Bt, int K) {
  int t = blockIdx.x * blockDim.x + threadIdx.x;
  if (t >= 128 * K) return;
  int n = t / K, k = t - n * K;
  Bt[t] = (bf16)w[(size_t)k * 128 + n];
}

// Sum the two K-split partials into columns [0..128) of A2 [R2 x 256].
__global__ void reduce_partials(const float* __restrict__ P, float* __restrict__ A2) {
  int t = blockIdx.x * blockDim.x + threadIdx.x; // R2_C*128 threads
  int r = t >> 7, c = t & 127;
  A2[(size_t)r * 256 + c] = P[t] + P[(size_t)R2_C * 128 + t];
}

// Convert float4[4] (16 f32 values in WMMA-A order) to a bf16 A fragment.
__device__ __forceinline__ v16bf cvt16(const float4* f) {
  v16bf av;
  av[0]  = (bf16)f[0].x; av[1]  = (bf16)f[0].y; av[2]  = (bf16)f[0].z; av[3]  = (bf16)f[0].w;
  av[4]  = (bf16)f[1].x; av[5]  = (bf16)f[1].y; av[6]  = (bf16)f[1].z; av[7]  = (bf16)f[1].w;
  av[8]  = (bf16)f[2].x; av[9]  = (bf16)f[2].y; av[10] = (bf16)f[2].z; av[11] = (bf16)f[2].w;
  av[12] = (bf16)f[3].x; av[13] = (bf16)f[3].y; av[14] = (bf16)f[3].z; av[15] = (bf16)f[3].w;
  return av;
}

// ---------------------------------------------------------------------------
// WMMA GEMM:  C[M x N=128] f32 = A[M x K] (f32 row-major, converted to bf16
// in-register) x B, where B is supplied pre-transposed as Bt[N=128 x K] bf16.
//
// Block = 256 threads = 8 waves; wave w owns N-tile w (cols 16w..16w+15).
// Each wave accumulates MT vertically-stacked 16x16 f32 tiles.
// K unrolled by 2: all loads first, then per phase convert-ALL then WMMA-ALL
// with both fragment arrays live, forcing distinct registers (no WMMA->VALU
// WAR v_nops) and keeping many loads in flight. +global_prefetch on A stream.
//
// Fragment layouts as loaded here (wave32; k contiguous per lane half, not ISA table order):
//   A 16x32 bf16: lane<16 holds row m, K=[k..k+16);
//                 lane>=16 holds K=[k+16..k+32)
//   B 32x16 bf16: lane<16 = col n, K=[k..k+16); lane>=16 = col n, K=[k+16..k+32)
//   C 16x16 f32 : VGPR g -> row g (lanes 0-15) / g+8 (lanes 16-31), col = lane%16
// ---------------------------------------------------------------------------
template <int MT>
__global__ __launch_bounds__(256) void gemm_wmma(
    const float* __restrict__ A, int lda,
    const bf16* __restrict__ Bt, int ldb,
    float* __restrict__ C, int ldc, size_t cSliceStride,
    int K, int relu) {
  const int lane = threadIdx.x & 31;
  const int wave = threadIdx.x >> 5;   // N-tile 0..7
  const int lm   = lane & 15;
  const int hs   = lane >> 4;          // half-select
  const int mbase = blockIdx.x * (MT * 16);
  const int n = wave * 16 + lm;

  const int span = K / gridDim.y;      // all spans divisible by 64
  const int kb = blockIdx.y * span;
  const int ke = kb + span;
  C += (size_t)blockIdx.y * cSliceStride;

  v8f acc[MT] = {};
  const bf16* bp = Bt + (size_t)n * ldb + hs * 16;
  const float* arow[MT];
#pragma unroll
  for (int t = 0; t < MT; ++t)
    arow[t] = A + (size_t)(mbase + t * 16 + lm) * lda + hs * 16;

  for (int k = kb; k < ke; k += 64) {
    // ---- issue ALL loads for two K-steps into distinct temps ----
    v16bf bv0 = *(const v16bf*)(bp + k);
    v16bf bv1 = *(const v16bf*)(bp + k + 32);
    float4 f0[MT][4];
    float4 f1[MT][4];
#pragma unroll
    for (int t = 0; t < MT; ++t) {
      const float* ap = arow[t] + k;
      f0[t][0] = *(const float4*)(ap);
      f0[t][1] = *(const float4*)(ap + 4);
      f0[t][2] = *(const float4*)(ap + 8);
      f0[t][3] = *(const float4*)(ap + 12);
    }
#pragma unroll
    for (int t = 0; t < MT; ++t) {
      const float* ap = arow[t] + k + 32;
      f1[t][0] = *(const float4*)(ap);
      f1[t][1] = *(const float4*)(ap + 4);
      f1[t][2] = *(const float4*)(ap + 8);
      f1[t][3] = *(const float4*)(ap + 12);
      // prefetch the linear A stream ~2KB / 16 K-steps ahead (speculative-safe)
      __builtin_prefetch((const void*)(ap + 512), 0, 0);
    }
    // ---- phase 0: convert ALL fragments, then WMMA ALL (distinct regs) ----
    v16bf av0[MT];
#pragma unroll
    for (int t = 0; t < MT; ++t) av0[t] = cvt16(f0[t]);
#pragma unroll
    for (int t = 0; t < MT; ++t)
      acc[t] = __builtin_amdgcn_wmma_f32_16x16x32_bf16(
          false, av0[t], false, bv0, (short)0, acc[t], false, false);
    // ---- phase 1: separate fragment array -> no WAR against phase-0 WMMAs ----
    v16bf av1[MT];
#pragma unroll
    for (int t = 0; t < MT; ++t) av1[t] = cvt16(f1[t]);
#pragma unroll
    for (int t = 0; t < MT; ++t)
      acc[t] = __builtin_amdgcn_wmma_f32_16x16x32_bf16(
          false, av1[t], false, bv1, (short)0, acc[t], false, false);
  }

#pragma unroll
  for (int t = 0; t < MT; ++t) {
#pragma unroll
    for (int g = 0; g < 8; ++g) {
      int row = mbase + t * 16 + g + hs * 8;
      float val = acc[t][g];
      if (relu) val = fmaxf(val, 0.0f);
      C[(size_t)row * ldc + wave * 16 + lm] = val;
    }
  }
}

// ---------------------------------------------------------------------------
// Per-row: L2-normalize x1[1024 x 128], run dense MLP 128->64->32->16->8->1,
// softmax over the singleton last axis (== 1.0 exactly).
// ---------------------------------------------------------------------------
__global__ void mlp_head(const float* __restrict__ X,
                         const float* __restrict__ wd1, const float* __restrict__ bd1,
                         const float* __restrict__ wd2, const float* __restrict__ bd2,
                         const float* __restrict__ wd3, const float* __restrict__ bd3,
                         const float* __restrict__ wd4, const float* __restrict__ bd4,
                         const float* __restrict__ wd5, const float* __restrict__ bd5,
                         float* __restrict__ out) {
  __shared__ float a[128];
  __shared__ float b[64];
  __shared__ float red[4];
  int r = blockIdx.x, tid = threadIdx.x; // 128 threads

  float v = X[(size_t)r * 128 + tid];
  float ss = v * v;
  for (int off = 16; off > 0; off >>= 1) ss += __shfl_down(ss, off, 32);
  if ((tid & 31) == 0) red[tid >> 5] = ss;
  __syncthreads();
  float tot = red[0] + red[1] + red[2] + red[3];
  a[tid] = v * rsqrtf(fmaxf(tot, 1e-12f));
  __syncthreads();

  if (tid < 64) {
    float acc = bd1[tid];
    for (int k = 0; k < 128; ++k) acc += a[k] * wd1[k * 64 + tid];
    b[tid] = fmaxf(acc, 0.0f);
  }
  __syncthreads();
  if (tid < 32) {
    float acc = bd2[tid];
    for (int k = 0; k < 64; ++k) acc += b[k] * wd2[k * 32 + tid];
    a[tid] = fmaxf(acc, 0.0f);
  }
  __syncthreads();
  if (tid < 16) {
    float acc = bd3[tid];
    for (int k = 0; k < 32; ++k) acc += a[k] * wd3[k * 16 + tid];
    b[tid] = fmaxf(acc, 0.0f);
  }
  __syncthreads();
  if (tid < 8) {
    float acc = bd4[tid];
    for (int k = 0; k < 16; ++k) acc += b[k] * wd4[k * 8 + tid];
    a[tid] = fmaxf(acc, 0.0f);
  }
  __syncthreads();
  if (tid == 0) {
    float z = bd5[0];
    for (int k = 0; k < 8; ++k) z += a[k] * wd5[k];
    // softmax over an axis of length 1: exp(z - max)/sum == 1.0 exactly
    out[r] = __expf(z - z);
  }
}

// ---------------------------------------------------------------------------
extern "C" void kernel_launch(void* const* d_in, const int* in_sizes, int n_in,
                              void* d_out, int out_size, void* d_ws, size_t ws_size,
                              hipStream_t stream) {
  (void)in_sizes; (void)n_in; (void)out_size; (void)ws_size;

  const int*   src_nodes = (const int*)d_in[0];
  const int*   s2src1    = (const int*)d_in[1];
  const int*   s2src2    = (const int*)d_in[2];
  const int*   s2dst1    = (const int*)d_in[3];
  const int*   s2dst2    = (const int*)d_in[4];
  const float* dif1      = (const float*)d_in[5];   // [R1 x S1]
  const float* dif2      = (const float*)d_in[6];   // [R2 x N2]
  const float* feat      = (const float*)d_in[7];   // [100000 x 128]
  const float* w1        = (const float*)d_in[8];   // [256 x 128]
  const float* w2        = (const float*)d_in[9];   // [256 x 128]
  const float* wd1 = (const float*)d_in[10]; const float* bd1 = (const float*)d_in[11];
  const float* wd2 = (const float*)d_in[12]; const float* bd2 = (const float*)d_in[13];
  const float* wd3 = (const float*)d_in[14]; const float* bd3 = (const float*)d_in[15];
  const float* wd4 = (const float*)d_in[16]; const float* bd4 = (const float*)d_in[17];
  const float* wd5 = (const float*)d_in[18]; const float* bd5 = (const float*)d_in[19];
  float* out = (float*)d_out;

  char* ws = (char*)d_ws;
  size_t off = 0;
  bf16*  Bt2  = (bf16*)(ws + off);  off += (size_t)128 * N2_C * 2;     // L2-resident B panel
  float* P    = (float*)(ws + off); off += (size_t)2 * R2_C * 128 * 4; // K-split partials
  float* A2   = (float*)(ws + off); off += (size_t)R2_C * 256 * 4;     // [agg2 | dst2]
  bf16*  Btw1 = (bf16*)(ws + off);  off += (size_t)128 * 256 * 2;
  bf16*  Btw2 = (bf16*)(ws + off);  off += (size_t)128 * 256 * 2;
  float* x2   = (float*)(ws + off); off += (size_t)R2_C * 128 * 4;     // hop-2 output
  bf16*  Bt3  = (bf16*)(ws + off);  off += (size_t)128 * S1_C * 2;
  float* A4   = (float*)(ws + off); off += (size_t)R1_C * 256 * 4;     // [agg1 | dst1]
  float* x1   = (float*)(ws + off); off += (size_t)R1_C * 128 * 4;     // hop-1 output

  // Stage 0: gathers + weight transposes (inputs only, independent)
  gather_bt<<<(N2_C + 255) / 256, 256, 0, stream>>>(s2src2, src_nodes, feat, Bt2, N2_C);
  gather_dst<<<R2_C, 32, 0, stream>>>(s2dst2, src_nodes, feat, A2);
  transpose_w<<<(128 * 256 + 255) / 256, 256, 0, stream>>>(w1, Btw1, 256);
  transpose_w<<<(128 * 256 + 255) / 256, 256, 0, stream>>>(w2, Btw2, 256);

  // Stage 1: agg2 = dif_mat_2 @ src  (M=4096, K=40960, N=128; K split in 2)
  gemm_wmma<4><<<dim3(R2_C / 64, 2), 256, 0, stream>>>(
      dif2, N2_C, Bt2, N2_C, P, 128, (size_t)R2_C * 128, N2_C, 0);
  reduce_partials<<<(R2_C * 128) / 256, 256, 0, stream>>>(P, A2);

  // Stage 2: x2 = relu([agg2|dst2] @ w1)  (M=4096, K=256, N=128)
  gemm_wmma<1><<<dim3(R2_C / 16, 1), 256, 0, stream>>>(
      A2, 256, Btw1, 256, x2, 128, 0, 256, 1);

  // Stage 3: hop-1 gathers from x2
  gather_bt<<<(S1_C + 255) / 256, 256, 0, stream>>>(s2src1, nullptr, x2, Bt3, S1_C);
  gather_dst<<<R1_C, 32, 0, stream>>>(s2dst1, nullptr, x2, A4);

  // Stage 4: agg1 = dif_mat_1 @ src1  (M=1024, K=8192, N=128) -> A4 cols [0,128)
  gemm_wmma<1><<<dim3(R1_C / 16, 1), 256, 0, stream>>>(
      dif1, S1_C, Bt3, S1_C, A4, 256, 0, S1_C, 0);

  // Stage 5: x1 = [agg1|dst1] @ w2  (no activation)
  gemm_wmma<1><<<dim3(R1_C / 16, 1), 256, 0, stream>>>(
      A4, 256, Btw2, 256, x1, 128, 0, 256, 0);

  // Stage 6: L2-normalize + dense head + singleton softmax
  mlp_head<<<R1_C, 128, 0, stream>>>(x1, wd1, bd1, wd2, bd2, wd3, bd3,
                                     wd4, bd4, wd5, bd5, out);
}
